// TreeLSTM_84189948936631
// MI455X (gfx1250) — matrix operand order repaired and checked by reading, not compiled, not run on hardware
//
#include <hip/hip_runtime.h>
#include <cstdint>
#include <cstddef>

// ---------------------------------------------------------------------------
// Tree-LSTM (4-ary, depth 8) for MI455X / gfx1250.
//  - bf16 WMMA (v_wmma_f32_16x16x32_bf16) for all GEMMs, f32 accumulation.
//  - One workgroup (8 waves / 256 threads, wave32) = 16-node tile per level.
//  - Only W_ih B-fragments stay register-resident (128 VGPRs); W_hh fragments
//    stream from L2 per step -> peak VGPRs < 256, no vgpr_msb shuffling.
//  - First WMMA of each chain uses inline-zero C; bias folded into gate math.
//  - Fast activations on v_exp_f32 / v_rcp_f32 instead of libm tanhf.
// ---------------------------------------------------------------------------

typedef __bf16 bf16_t;
typedef __attribute__((ext_vector_type(16))) __bf16 v16bf;
typedef __attribute__((ext_vector_type(8)))  float  v8f;

#define TREE_DEPTH 8
#define NN   87381          // (4^9 - 1)/3 nodes
#define EV   128            // embedding dim
#define HV   128            // hidden dim
#define G4   512            // 4*H gate width
#define NB   16             // nodes per workgroup tile
#define TMAX 6

union FragU { uint32_t u[8]; uint4 q[2]; v16bf v; };

__device__ __forceinline__ float fast_rcp(float x) { return __builtin_amdgcn_rcpf(x); }
__device__ __forceinline__ float sigm(float x)     { return fast_rcp(1.0f + __expf(-x)); }
// tanh(x) = 1 - 2/(e^{2x}+1); saturates correctly to +/-1 for large |x|.
__device__ __forceinline__ float tanh_fast(float x){ return 1.0f - 2.0f * fast_rcp(1.0f + __expf(2.0f * x)); }

// A fragment (16x32 bf16), rows m=0..15 from row-major src (ld elems), k window k0.
// ISA 7.12.2: lanes 0-15 hold K={0..7,16..23}, lanes 16-31 K={8..15,24..31}, K-pairs/dword.
__device__ __forceinline__ v16bf load_a_frag(const bf16_t* __restrict__ src,
                                             int ld, int k0, int lane) {
    FragU f;
    int m  = lane & 15;
    int kb = (lane >> 4) << 3;             // 0 or 8
    const bf16_t* p = src + m * ld + k0 + kb;
#pragma unroll
    for (int v = 0; v < 8; ++v) {
        int k = (v < 4) ? (2 * v) : (16 + 2 * (v - 4));
        f.u[v] = *(const uint32_t*)(p + k);
    }
    return f.v;
}

// B fragment (32x16 bf16) from K-major weights wt[k][n] (row stride ld elems):
// lane holds column n = lane%16, K in same order as A fragment (scalar loads).
__device__ __forceinline__ v16bf load_b_frag(const bf16_t* __restrict__ wt,
                                             int ld, int k0, int n0, int lane) {
    FragU f;
    int kb = (lane >> 4) << 3;
    const bf16_t* p = wt + (size_t)k0 * ld + n0 + (lane & 15);
#pragma unroll
    for (int s = 0; s < 16; ++s) {
        int k = (s < 8) ? (kb + s) : (16 + kb + (s - 8));
        f.v[s] = p[(size_t)k * ld];
    }
    return f.v;
}

// f32 (rows,cols) row-major -> bf16 (cols,rows) row-major (K-major for WMMA B).
__global__ void transpose_bf16_kernel(const float* __restrict__ in,
                                      bf16_t* __restrict__ out,
                                      int rows, int cols) {
    int i = blockIdx.x * blockDim.x + threadIdx.x;
    if (i < rows * cols) {
        int r = i / cols, c = i - r * cols;
        out[c * rows + r] = (bf16_t)in[i];
    }
}

__global__ __launch_bounds__(256)
void tree_level_kernel(const int* __restrict__ labels,
                       const float* __restrict__ emb,
                       const bf16_t* __restrict__ wih_f_t,   // 128x512 K-major
                       const bf16_t* __restrict__ whh_f_t,   // 128x512
                       const bf16_t* __restrict__ wih_b_t,
                       const bf16_t* __restrict__ whh_b_t,
                       const float* __restrict__ b_ih_f, const float* __restrict__ b_hh_f,
                       const float* __restrict__ b_ih_b, const float* __restrict__ b_hh_b,
                       const bf16_t* __restrict__ lin_t,     // 256x128 K-major
                       const float* __restrict__ lin_b,
                       bf16_t* __restrict__ enc,
                       float* __restrict__ out0,
                       int level, int start, int count, int cstart, int T) {
    __shared__ bf16_t s_seq[TMAX * NB * EV];   // 24 KB: sequence tile
    __shared__ bf16_t s_h[NB * HV];            //  4 KB: recurrent h (bf16)
    __shared__ bf16_t s_cat[NB * 2 * HV];      //  8 KB: [h_f | h_b]

    const int tid   = threadIdx.x;
    const int lane  = tid & 31;
    const int wave  = tid >> 5;                // 0..7
    const int tile0 = blockIdx.x * NB;
    const int n_local = wave * 16 + (lane & 15);   // H column owned by this lane
    const int mhi     = (lane >> 4) << 3;          // row offset for D-layout

    // ---- stage sequence tile into LDS (x rows; children rows if interior) ----
    for (int idx = tid; idx < NB * EV; idx += blockDim.x) {
        int m = idx >> 7, e = idx & 127;
        int node = tile0 + m; if (node >= count) node = count - 1;
        bf16_t v = (bf16_t)emb[(size_t)labels[start + node] * EV + e];
        s_seq[idx] = v;                       // t = 0
        s_seq[(T - 1) * NB * EV + idx] = v;   // t = T-1
    }
    if (T > 2) {
        for (int idx = tid; idx < 4 * NB * EV; idx += blockDim.x) {
            int a = idx / (NB * EV);
            int r = idx - a * NB * EV;
            int m = r >> 7, e = r & 127;
            int node = tile0 + m; if (node >= count) node = count - 1;
            s_seq[(1 + a) * NB * EV + r] =
                enc[(size_t)(cstart + node * 4 + a) * EV + e];
        }
    }
    __syncthreads();

    const v8f zc = {};   // inline-zero C matrix

    for (int dir = 0; dir < 2; ++dir) {
        const bf16_t* gwih = dir ? wih_b_t : wih_f_t;
        const bf16_t* gwhh = dir ? whh_b_t : whh_f_t;
        const float*  bi   = dir ? b_ih_b  : b_ih_f;
        const float*  bh   = dir ? b_hh_b  : b_hh_f;

        // Register-resident W_ih fragments (used every timestep): 16 x v16bf.
        v16bf fih[4][4];
#pragma unroll
        for (int kb = 0; kb < 4; ++kb)
#pragma unroll
            for (int j = 0; j < 4; ++j)
                fih[kb][j] = load_b_frag(gwih, G4, kb * 32, j * 128 + wave * 16, lane);

        float bias[4];
#pragma unroll
        for (int j = 0; j < 4; ++j)
            bias[j] = bi[j * 128 + n_local] + bh[j * 128 + n_local];

        v8f c = {};   // per-lane c tile (D-layout), persists in registers

        for (int step = 0; step < T; ++step) {
            const int t = dir ? (T - 1 - step) : step;
            const bf16_t* xrow = s_seq + t * NB * EV;
            v8f acc[4];

            // g = x_t @ W_ih^T   (first WMMA with zero C; B register-resident)
            {
                v16bf af0 = load_a_frag(xrow, EV, 0, lane);
#pragma unroll
                for (int j = 0; j < 4; ++j)
                    acc[j] = __builtin_amdgcn_wmma_f32_16x16x32_bf16(
                        false, af0, false, fih[0][j], (short)0, zc, false, false);
            }
#pragma unroll
            for (int kb = 1; kb < 4; ++kb) {
                v16bf af = load_a_frag(xrow, EV, kb * 32, lane);
#pragma unroll
                for (int j = 0; j < 4; ++j)
                    acc[j] = __builtin_amdgcn_wmma_f32_16x16x32_bf16(
                        false, af, false, fih[kb][j], (short)0, acc[j], false, false);
            }
            // g += h @ W_hh^T  (h == 0 at step 0 -> skip; B streamed from L2)
            if (step > 0) {
#pragma unroll
                for (int kb = 0; kb < 4; ++kb) {
                    v16bf af = load_a_frag(s_h, HV, kb * 32, lane);
#pragma unroll
                    for (int j = 0; j < 4; ++j) {
                        v16bf bw = load_b_frag(gwhh, G4, kb * 32, j * 128 + wave * 16, lane);
                        acc[j] = __builtin_amdgcn_wmma_f32_16x16x32_bf16(
                            false, af, false, bw, (short)0, acc[j], false, false);
                    }
                }
            }

            __syncthreads();   // all waves done reading old s_h
            // gates: i,f,g,o = acc0..3 + bias (lane-local, D-layout aligned)
#pragma unroll
            for (int r = 0; r < 8; ++r) {
                float gi = acc[0][r] + bias[0];
                float gf = acc[1][r] + bias[1];
                float gg = acc[2][r] + bias[2];
                float go = acc[3][r] + bias[3];
                float cn = sigm(gf) * c[r] + sigm(gi) * tanh_fast(gg);
                c[r] = cn;
                float hn = sigm(go) * tanh_fast(cn);
                int m = mhi + r;
                s_h[m * HV + n_local] = (bf16_t)hn;
                if (step == T - 1)
                    s_cat[m * (2 * HV) + dir * HV + n_local] = (bf16_t)hn;
            }
            __syncthreads();   // new s_h (and s_cat on last step) visible
        }
    }

    // ---- out = tanh([h_f|h_b] @ lin_w^T + lin_b); wave w -> cols 16w..16w+16
    {
        v8f acc;
        {
            v16bf af0 = load_a_frag(s_cat, 2 * HV, 0, lane);
            v16bf bf0 = load_b_frag(lin_t, EV, 0, wave * 16, lane);
            acc = __builtin_amdgcn_wmma_f32_16x16x32_bf16(
                false, af0, false, bf0, (short)0, zc, false, false);
        }
#pragma unroll
        for (int kb = 1; kb < 8; ++kb) {
            v16bf af = load_a_frag(s_cat, 2 * HV, kb * 32, lane);
            v16bf bw = load_b_frag(lin_t, EV, kb * 32, wave * 16, lane);
            acc = __builtin_amdgcn_wmma_f32_16x16x32_bf16(
                false, af, false, bw, (short)0, acc, false, false);
        }
        float bl = lin_b[n_local];
#pragma unroll
        for (int r = 0; r < 8; ++r) {
            float v = tanh_fast(acc[r] + bl);
            int m = mhi + r;
            int node = tile0 + m;
            if (node < count) {
                enc[(size_t)(start + node) * EV + n_local] = (bf16_t)v;
                if (level == 0 && node == 0) out0[n_local] = v;  // root row, f32
            }
        }
    }
}

extern "C" void kernel_launch(void* const* d_in, const int* in_sizes, int n_in,
                              void* d_out, int out_size, void* d_ws, size_t ws_size,
                              hipStream_t stream) {
    (void)in_sizes; (void)n_in; (void)out_size; (void)ws_size;

    const int*   labels = (const int*)  d_in[0];
    const float* emb    = (const float*)d_in[1];
    const float* w_ih_f = (const float*)d_in[2];
    const float* w_hh_f = (const float*)d_in[3];
    const float* b_ih_f = (const float*)d_in[4];
    const float* b_hh_f = (const float*)d_in[5];
    const float* w_ih_b = (const float*)d_in[6];
    const float* w_hh_b = (const float*)d_in[7];
    const float* b_ih_b = (const float*)d_in[8];
    const float* b_hh_b = (const float*)d_in[9];
    const float* lin_w  = (const float*)d_in[10];
    const float* lin_b  = (const float*)d_in[11];

    // Workspace layout (bytes); total ~23 MB.
    char* ws = (char*)d_ws;
    bf16_t* wih_f_t = (bf16_t*)(ws + 0);        // 128x512 bf16 (128 KB)
    bf16_t* whh_f_t = (bf16_t*)(ws + 131072);
    bf16_t* wih_b_t = (bf16_t*)(ws + 262144);
    bf16_t* whh_b_t = (bf16_t*)(ws + 393216);
    bf16_t* lin_t   = (bf16_t*)(ws + 524288);   // 256x128 bf16 (64 KB)
    bf16_t* enc     = (bf16_t*)(ws + 589824);   // NN x 128 bf16 (~21.3 MB)

    // Prep: transpose + f32->bf16 (K-major layout for WMMA B fragments).
    transpose_bf16_kernel<<<(512 * 128 + 255) / 256, 256, 0, stream>>>(w_ih_f, wih_f_t, 512, 128);
    transpose_bf16_kernel<<<(512 * 128 + 255) / 256, 256, 0, stream>>>(w_hh_f, whh_f_t, 512, 128);
    transpose_bf16_kernel<<<(512 * 128 + 255) / 256, 256, 0, stream>>>(w_ih_b, wih_b_t, 512, 128);
    transpose_bf16_kernel<<<(512 * 128 + 255) / 256, 256, 0, stream>>>(w_hh_b, whh_b_t, 512, 128);
    transpose_bf16_kernel<<<(128 * 256 + 255) / 256, 256, 0, stream>>>(lin_w, lin_t, 128, 256);

    // Level offsets: (4^l - 1)/3.
    int offs[TREE_DEPTH + 2];
    offs[0] = 0;
    for (int l = 1; l <= TREE_DEPTH + 1; ++l)
        offs[l] = offs[l - 1] + (1 << (2 * (l - 1)));

    for (int level = TREE_DEPTH; level >= 0; --level) {
        int start  = offs[level];
        int count  = 1 << (2 * level);
        int cstart = (level < TREE_DEPTH) ? offs[level + 1] : 0;
        int T      = (level == TREE_DEPTH) ? 2 : TMAX;
        int nwg    = (count + NB - 1) / NB;
        tree_level_kernel<<<nwg, 256, 0, stream>>>(
            labels, emb, wih_f_t, whh_f_t, wih_b_t, whh_b_t,
            b_ih_f, b_hh_f, b_ih_b, b_hh_b, lin_t, lin_b,
            enc, (float*)d_out, level, start, count, cstart, T);
    }
}
